// Attention_27608049779174
// MI455X (gfx1250) — matrix operand order repaired and checked by reading, not compiled, not run on hardware
//
#include <hip/hip_runtime.h>
#include <hip/hip_bf16.h>

typedef __attribute__((ext_vector_type(16))) _Float16 v16h;
typedef __attribute__((ext_vector_type(8)))  _Float16 v8h;
typedef __attribute__((ext_vector_type(8)))  float    v8f;

union Frag { v16h v; v8h h[2]; };

__device__ inline v8f wmma_f(const Frag& a, const Frag& b, v8f c) {
    return __builtin_amdgcn_wmma_f32_16x16x32_f16(
        /*neg_a=*/false, a.v, /*neg_b=*/false, b.v,
        /*c_mod=*/(short)0, c, /*reuse_a=*/false, /*reuse_b=*/false);
}

__device__ inline v8f zero8() {
    v8f z = {0.f, 0.f, 0.f, 0.f, 0.f, 0.f, 0.f, 0.f};
    return z;
}

#define DIMC  256
#define SPB   16384        // 128*128 spatial per batch image
#define NWIN  2048         // 4 * 8 heads * 8 * 8 windows
#define WREG  ((size_t)NWIN * 16384)   // halfs per q/k/v region (win*256tok*64d)
#define QOFF  ((size_t)0)
#define KOFF  (WREG)
#define VOFF  (2 * WREG)
#define AOOFF (3 * WREG)
#define SCALE 0.125f       // 64^-0.5

// ---------------------------------------------------------------------------
// Kernel 1: fused Q/K/V projection, software-pipelined.
// out[o,s] = sum_c W[o,c]*x[c,s]; o<512 -> Q(scaled), <1024 -> K, else V.
// Double-buffered LDS tiles: issue next chunk's global loads, run WMMA chain,
// then convert+store to the alternate buffer; one barrier per K-chunk.
// ---------------------------------------------------------------------------
__global__ __launch_bounds__(256) void qkv_proj(
    const float* __restrict__ x, const float* __restrict__ Wq,
    const float* __restrict__ Wkv, _Float16* __restrict__ ws)
{
    __shared__ _Float16 Ah[2][128 * 32];   // [buf][o_local][c]  2x8 KB
    __shared__ _Float16 Bh[2][128 * 32];   // [buf][s_local][c]  2x8 KB

    const int t = threadIdx.x;
    const int lane = t & 31, wave = t >> 5;
    const int lm = lane & 15, lh = lane >> 4;
    const int sBase = blockIdx.x * 128;
    const int oBase = blockIdx.y * 128;
    const int bb  = sBase / SPB;
    const int hw0 = sBase % SPB;

    const int ca = t & 31, ol = t >> 5;     // A staging coords
    const int sl = t & 127, ch = t >> 7;    // B staging coords

    v8f acc[8];
#pragma unroll
    for (int i = 0; i < 8; ++i) acc[i] = zero8();

    float aReg[16], bReg[16];

    // prologue: stage chunk 0 into buffer 0
    {
#pragma unroll
        for (int r = 0; r < 16; ++r) {
            const int og = oBase + ol + r * 8;
            aReg[r] = (og < 512) ? Wq[og * DIMC + ca]
                                 : Wkv[(og - 512) * DIMC + ca];
        }
#pragma unroll
        for (int i = 0; i < 16; ++i)
            bReg[i] = x[(size_t)(bb * DIMC + ch + i * 2) * SPB + hw0 + sl];
#pragma unroll
        for (int r = 0; r < 16; ++r) Ah[0][(ol + r * 8) * 32 + ca] = (_Float16)aReg[r];
#pragma unroll
        for (int i = 0; i < 16; ++i) Bh[0][sl * 32 + ch + i * 2] = (_Float16)bReg[i];
    }
    __syncthreads();

    for (int kc = 0; kc < 8; ++kc) {
        const int buf = kc & 1;
        const bool more = (kc + 1 < 8);
        // --- issue next chunk's global loads (results consumed after WMMAs) ---
        if (more) {
            const int c0n = (kc + 1) * 32;
#pragma unroll
            for (int r = 0; r < 16; ++r) {
                const int og = oBase + ol + r * 8;
                aReg[r] = (og < 512) ? Wq[og * DIMC + c0n + ca]
                                     : Wkv[(og - 512) * DIMC + c0n + ca];
            }
#pragma unroll
            for (int i = 0; i < 16; ++i)
                bReg[i] = x[(size_t)(bb * DIMC + c0n + ch + i * 2) * SPB + hw0 + sl];
            if (kc + 2 < 8)   // speculative hint for the chunk after next
                __builtin_prefetch(&x[(size_t)(bb * DIMC + (kc + 2) * 32 + ch) * SPB + hw0 + sl], 0, 1);
        }
        // --- WMMA chain on current buffer, fragments batched in groups of 4 ---
        Frag af;
        {
            const int row = wave * 16 + lm;
            af.h[0] = *(const v8h*)&Ah[buf][row * 32 + lh * 8];
            af.h[1] = *(const v8h*)&Ah[buf][row * 32 + 16 + lh * 8];
        }
#pragma unroll
        for (int g = 0; g < 2; ++g) {
            Frag bf[4];
#pragma unroll
            for (int i = 0; i < 4; ++i) {
                const int srow = (g * 4 + i) * 16 + lm;
                bf[i].h[0] = *(const v8h*)&Bh[buf][srow * 32 + lh * 8];
                bf[i].h[1] = *(const v8h*)&Bh[buf][srow * 32 + 16 + lh * 8];
            }
#pragma unroll
            for (int i = 0; i < 4; ++i)
                acc[g * 4 + i] = wmma_f(af, bf[i], acc[g * 4 + i]);
        }
        // --- convert + store next chunk into alternate buffer ---
        if (more) {
#pragma unroll
            for (int r = 0; r < 16; ++r)
                Ah[buf ^ 1][(ol + r * 8) * 32 + ca] = (_Float16)aReg[r];
#pragma unroll
            for (int i = 0; i < 16; ++i)
                Bh[buf ^ 1][sl * 32 + ch + i * 2] = (_Float16)bReg[i];
        }
        __syncthreads();
    }

    // --- epilogue: scatter to windowed q/k/v (f16) ---
#pragma unroll
    for (int nt = 0; nt < 8; ++nt) {
        const int sg = sBase + nt * 16 + lm;
        const int hw = sg % SPB;
        const int hh = hw >> 7, wwc = hw & 127;
        const int X = hh >> 4, Y = wwc >> 4;
        const int tok = (hh & 15) * 16 + (wwc & 15);
#pragma unroll
        for (int r = 0; r < 8; ++r) {
            const int og = oBase + wave * 16 + r + 8 * lh;
            float v = acc[nt][r];
            const int head = (og & 511) >> 6;
            const int dd = og & 63;
            size_t base;
            if (og < 512)       { base = QOFF; v *= SCALE; }
            else if (og < 1024) { base = KOFF; }
            else                { base = VOFF; }
            const int widw = ((bb * 8 + head) * 8 + X) * 8 + Y;
            ws[base + (size_t)widw * 16384 + tok * 64 + dd] = (_Float16)v;
        }
    }
}

// ---------------------------------------------------------------------------
// Kernel 2: per-window flash attention, one block per window.
// 8 waves, each owns 32 query rows (two 16-row C-tiles, f32 accumulators).
// K-fragments are row-tile independent -> loaded once per 32-key step;
// V-fragments likewise. Vt and Pl are ping-ponged: one barrier per step.
// ---------------------------------------------------------------------------
__global__ __launch_bounds__(256) void win_attn(_Float16* __restrict__ ws)
{
    __shared__ _Float16 Vt[2][64 * 32];         // [buf][d][key]     2x4 KB
    __shared__ _Float16 Pl[2][8 * 2 * 16 * 32]; // [buf][(wave,tt)]  2x16 KB

    const int t = threadIdx.x, lane = t & 31, wave = t >> 5;
    const int lm = lane & 15, lh = lane >> 4;
    const int win = blockIdx.x;

    const _Float16* qg = ws + QOFF + (size_t)win * 16384;
    const _Float16* kg = ws + KOFF + (size_t)win * 16384;
    const _Float16* vg = ws + VOFF + (size_t)win * 16384;
    _Float16* outw     = ws + AOOFF + (size_t)win * 16384;

    const int q0 = wave * 32;

    // q fragments (A-layout), resident across the whole loop
    Frag qf[2][2];
#pragma unroll
    for (int tt = 0; tt < 2; ++tt) {
        const int row = q0 + tt * 16 + lm;
#pragma unroll
        for (int kk = 0; kk < 2; ++kk) {
            qf[tt][kk].h[0] = *(const v8h*)&qg[row * 64 + kk * 32 + lh * 8];
            qf[tt][kk].h[1] = *(const v8h*)&qg[row * 64 + kk * 32 + 16 + lh * 8];
        }
    }

    v8f acc[2][4];
    float mrow[2][8], srow[2][8];
#pragma unroll
    for (int tt = 0; tt < 2; ++tt) {
#pragma unroll
        for (int dt = 0; dt < 4; ++dt) acc[tt][dt] = zero8();
#pragma unroll
        for (int r = 0; r < 8; ++r) { mrow[tt][r] = -3.0e38f; srow[tt][r] = 0.f; }
    }

    const int vd = t & 63, vkp = t >> 6;   // V staging coords

    for (int jt = 0; jt < 8; ++jt) {
        const int j0 = jt * 32;
        const int buf = jt & 1;
        // --- stage V transposed into ping-pong buffer ---
#pragma unroll
        for (int p = 0; p < 8; ++p) {
            const int key = p * 4 + vkp;
            Vt[buf][vd * 32 + key] = vg[(j0 + key) * 64 + vd];
        }
        // --- K fragments: shared by both row tiles, batched global loads ---
        Frag kf[2][2];   // [16-token group][d-chunk]
#pragma unroll
        for (int g = 0; g < 2; ++g) {
            const int tok = j0 + g * 16 + lm;
#pragma unroll
            for (int kk = 0; kk < 2; ++kk) {
                kf[g][kk].h[0] = *(const v8h*)&kg[tok * 64 + kk * 32 + lh * 8];
                kf[g][kk].h[1] = *(const v8h*)&kg[tok * 64 + kk * 32 + 16 + lh * 8];
            }
        }
        // --- sim + online softmax per row tile ---
#pragma unroll
        for (int tt = 0; tt < 2; ++tt) {
            v8f s0 = zero8(), s1 = zero8();
            s0 = wmma_f(qf[tt][0], kf[0][0], s0);
            s0 = wmma_f(qf[tt][1], kf[0][1], s0);
            s1 = wmma_f(qf[tt][0], kf[1][0], s1);
            s1 = wmma_f(qf[tt][1], kf[1][1], s1);
#pragma unroll
            for (int r = 0; r < 8; ++r) {
                float mt = fmaxf(s0[r], s1[r]);
                mt = fmaxf(mt, __shfl_xor(mt, 1, 32));
                mt = fmaxf(mt, __shfl_xor(mt, 2, 32));
                mt = fmaxf(mt, __shfl_xor(mt, 4, 32));
                mt = fmaxf(mt, __shfl_xor(mt, 8, 32));   // stays within lane-half
                const float mn = fmaxf(mrow[tt][r], mt);
                const float corr = __expf(mrow[tt][r] - mn);
                const float p0 = __expf(s0[r] - mn);
                const float p1 = __expf(s1[r] - mn);
                float rs = p0 + p1;
                rs += __shfl_xor(rs, 1, 32);
                rs += __shfl_xor(rs, 2, 32);
                rs += __shfl_xor(rs, 4, 32);
                rs += __shfl_xor(rs, 8, 32);
                srow[tt][r] = srow[tt][r] * corr + rs;
                mrow[tt][r] = mn;
#pragma unroll
                for (int dt = 0; dt < 4; ++dt) acc[tt][dt][r] *= corr;
                const int prow = (wave * 2 + tt) * 16 + r + 8 * lh;
                Pl[buf][prow * 32 + lm]      = (_Float16)p0;
                Pl[buf][prow * 32 + 16 + lm] = (_Float16)p1;
            }
        }
        __syncthreads();   // Vt[buf] + Pl[buf] visible; single barrier per step
        // --- out += P * V ; V fragments shared by both row tiles ---
        Frag vf[4];
#pragma unroll
        for (int dt = 0; dt < 4; ++dt) {
            vf[dt].h[0] = *(const v8h*)&Vt[buf][(dt * 16 + lm) * 32 + lh * 8];
            vf[dt].h[1] = *(const v8h*)&Vt[buf][(dt * 16 + lm) * 32 + 16 + lh * 8];
        }
#pragma unroll
        for (int tt = 0; tt < 2; ++tt) {
            Frag pf;
            const int prow = (wave * 2 + tt) * 16 + lm;
            pf.h[0] = *(const v8h*)&Pl[buf][prow * 32 + lh * 8];
            pf.h[1] = *(const v8h*)&Pl[buf][prow * 32 + 16 + lh * 8];
#pragma unroll
            for (int dt = 0; dt < 4; ++dt)
                acc[tt][dt] = wmma_f(pf, vf[dt], acc[tt][dt]);
        }
    }

    // --- finalize: divide by row sums, store windowed attention output ---
#pragma unroll
    for (int tt = 0; tt < 2; ++tt) {
#pragma unroll
        for (int r = 0; r < 8; ++r) {
            const float inv = 1.0f / srow[tt][r];
            const int row = q0 + tt * 16 + r + 8 * lh;
#pragma unroll
            for (int dt = 0; dt < 4; ++dt) {
                outw[row * 64 + dt * 16 + lm] = (_Float16)(acc[tt][dt][r] * inv);
            }
        }
    }
}

// ---------------------------------------------------------------------------
// Kernel 3: output projection, software-pipelined like kernel 1.
// out[o,s] = sum_c2 Wo[o,c2]*ao[c2,s] + bo[o]; gathers windowed ao via LDS.
// ---------------------------------------------------------------------------
__global__ __launch_bounds__(256) void out_proj(
    const float* __restrict__ Wo, const float* __restrict__ bo,
    const _Float16* __restrict__ ws, float* __restrict__ out)
{
    __shared__ _Float16 Ah[2][128 * 32];
    __shared__ _Float16 Bh[2][128 * 32];

    const int t = threadIdx.x;
    const int lane = t & 31, wave = t >> 5;
    const int lm = lane & 15, lh = lane >> 4;
    const int sBase = blockIdx.x * 128;
    const int oBase = blockIdx.y * 128;
    const int bb  = sBase / SPB;
    const int hw0 = sBase % SPB;
    const int hh  = hw0 >> 7;          // image row; columns = s_local
    const int X = hh >> 4, w1 = hh & 15;

    const int ca = t & 31, ol = t >> 5;
    const int sl = t & 127, ch = t >> 7;
    const int Y = sl >> 4, w2 = sl & 15;
    const int tok = w1 * 16 + w2;

    v8f acc[8];
#pragma unroll
    for (int i = 0; i < 8; ++i) acc[i] = zero8();

    float aReg[16];
    _Float16 bRegH[16];

    // prologue: stage chunk 0
    {
#pragma unroll
        for (int r = 0; r < 16; ++r)
            aReg[r] = Wo[(oBase + ol + r * 8) * 512 + ca];
#pragma unroll
        for (int i = 0; i < 16; ++i) {
            const int c2 = ch + i * 2;
            const int head = c2 >> 6, dd = c2 & 63;
            const int widw = ((bb * 8 + head) * 8 + X) * 8 + Y;
            bRegH[i] = ws[AOOFF + (size_t)widw * 16384 + tok * 64 + dd];
        }
#pragma unroll
        for (int r = 0; r < 16; ++r) Ah[0][(ol + r * 8) * 32 + ca] = (_Float16)aReg[r];
#pragma unroll
        for (int i = 0; i < 16; ++i) Bh[0][sl * 32 + ch + i * 2] = bRegH[i];
    }
    __syncthreads();

    for (int kc = 0; kc < 16; ++kc) {
        const int buf = kc & 1;
        const bool more = (kc + 1 < 16);
        if (more) {
            const int c0n = (kc + 1) * 32;
#pragma unroll
            for (int r = 0; r < 16; ++r)
                aReg[r] = Wo[(oBase + ol + r * 8) * 512 + c0n + ca];
#pragma unroll
            for (int i = 0; i < 16; ++i) {
                const int c2 = c0n + ch + i * 2;
                const int head = c2 >> 6, dd = c2 & 63;
                const int widw = ((bb * 8 + head) * 8 + X) * 8 + Y;
                bRegH[i] = ws[AOOFF + (size_t)widw * 16384 + tok * 64 + dd];
            }
        }
        Frag af;
        {
            const int row = wave * 16 + lm;
            af.h[0] = *(const v8h*)&Ah[buf][row * 32 + lh * 8];
            af.h[1] = *(const v8h*)&Ah[buf][row * 32 + 16 + lh * 8];
        }
#pragma unroll
        for (int g = 0; g < 2; ++g) {
            Frag bf[4];
#pragma unroll
            for (int i = 0; i < 4; ++i) {
                const int srow = (g * 4 + i) * 16 + lm;
                bf[i].h[0] = *(const v8h*)&Bh[buf][srow * 32 + lh * 8];
                bf[i].h[1] = *(const v8h*)&Bh[buf][srow * 32 + 16 + lh * 8];
            }
#pragma unroll
            for (int i = 0; i < 4; ++i)
                acc[g * 4 + i] = wmma_f(af, bf[i], acc[g * 4 + i]);
        }
        if (more) {
#pragma unroll
            for (int r = 0; r < 16; ++r)
                Ah[buf ^ 1][(ol + r * 8) * 32 + ca] = (_Float16)aReg[r];
#pragma unroll
            for (int i = 0; i < 16; ++i)
                Bh[buf ^ 1][sl * 32 + ch + i * 2] = bRegH[i];
        }
        __syncthreads();
    }

#pragma unroll
    for (int nt = 0; nt < 8; ++nt) {
        const int sg = sBase + nt * 16 + lm;
        const int hw = sg % SPB;
#pragma unroll
        for (int r = 0; r < 8; ++r) {
            const int og = oBase + wave * 16 + r + 8 * lh;
            out[((size_t)bb * 256 + og) * SPB + hw] = acc[nt][r] + bo[og];
        }
    }
}

extern "C" void kernel_launch(void* const* d_in, const int* in_sizes, int n_in,
                              void* d_out, int out_size, void* d_ws, size_t ws_size,
                              hipStream_t stream) {
    const float* x   = (const float*)d_in[0];
    const float* Wq  = (const float*)d_in[1];
    const float* Wkv = (const float*)d_in[2];
    const float* Wo  = (const float*)d_in[3];
    const float* bo  = (const float*)d_in[4];
    float* out = (float*)d_out;
    _Float16* ws = (_Float16*)d_ws;

    dim3 g1(512, 12);                 // 65536/128 s-tiles x 1536/128 o-tiles
    qkv_proj<<<g1, dim3(256), 0, stream>>>(x, Wq, Wkv, ws);

    win_attn<<<dim3(NWIN), dim3(256), 0, stream>>>(ws);

    dim3 g3(512, 2);                  // 65536/128 s-tiles x 256/128 o-tiles
    out_proj<<<g3, dim3(256), 0, stream>>>(Wo, bo, ws, out);
}
